// JRWindowModel_42958262894928
// MI455X (gfx1250) — matrix operand order repaired and checked by reading, not compiled, not run on hardware
//
#include <hip/hip_runtime.h>
#include <hip/hip_bf16.h>
#include <math.h>

typedef __attribute__((ext_vector_type(16))) _Float16 v16h;
typedef __attribute__((ext_vector_type(8)))  float    v8f;

#define NN 4096
#define JR_DT 1e-4f
#define JR_IN_CLAMP 500.0f
#define JR_ST_CLAMP 1000.0f

__device__ __forceinline__ float boundedf(float x, float c) { return c * tanhf(x / c); }

// ---------------------------------------------------------------------------
// Prep: Wsym = 0.5*exp(w_gain)*(sc + sc^T), packed into CDNA5 WMMA A-fragment
// layout (16x32 f16 tiles), plus per-row rowsum and sum-of-squares partials.
// One block per matrix row; deterministic fixed-order LDS reductions.
// ---------------------------------------------------------------------------
__global__ void __launch_bounds__(256) jr_prep_w(const float* __restrict__ sc,
                                                 const float* __restrict__ params,
                                                 _Float16* __restrict__ packedW,
                                                 float* __restrict__ rowsum,
                                                 float* __restrict__ sqpart)
{
    __shared__ float s_rs[256];
    __shared__ float s_ss[256];
    const int i   = blockIdx.x;        // matrix row
    const int tid = threadIdx.x;       // 256 threads
    const float hg = 0.5f * expf(params[16]);   // 0.5 * exp(w_gain)
    const int m = i >> 4;              // row tile
    const int r = i & 15;              // row within tile

    float rs = 0.f, ssq = 0.f;
    #pragma unroll
    for (int it = 0; it < 16; ++it) {
        const int j = tid + (it << 8);
        const float w = hg * (sc[(size_t)i * NN + j] + sc[(size_t)j * NN + i]);
        rs  += w;
        ssq += w * w;
        // A-fragment position in this code's contiguous K order (16-bit A 16x32):
        //  lane<16 : K {0..15}; lane>=16 : K {16..31}
        const int kb   = j >> 5;
        const int kk   = j & 31;
        const int lane = r + ((kk >> 4) << 4);
        const int e    = kk & 15;
        packedW[((size_t)(m * 128 + kb)) * 512 + lane * 16 + e] = (_Float16)w;
    }
    s_rs[tid] = rs; s_ss[tid] = ssq;
    __syncthreads();
    for (int off = 128; off; off >>= 1) {
        if (tid < off) { s_rs[tid] += s_rs[tid + off]; s_ss[tid] += s_ss[tid + off]; }
        __syncthreads();
    }
    if (tid == 0) { rowsum[i] = s_rs[0]; sqpart[i] = s_ss[0]; }
}

// Finalize Frobenius norm -> inv_norm scalar (deterministic single-block sum).
__global__ void jr_norm(const float* __restrict__ sqpart, float* __restrict__ scal)
{
    __shared__ float s[256];
    const int tid = threadIdx.x;
    float acc = 0.f;
    #pragma unroll
    for (int k = 0; k < 16; ++k) acc += sqpart[tid + 256 * k];
    s[tid] = acc;
    __syncthreads();
    for (int off = 128; off; off >>= 1) {
        if (tid < off) s[tid] += s[tid + off];
        __syncthreads();
    }
    if (tid == 0) scal[0] = 1.0f / sqrtf(s[0]);
}

// Initialize double-buffered state + f16 copy of P.
__global__ void jr_init(const float* __restrict__ P0, const float* __restrict__ E0,
                        const float* __restrict__ I0, const float* __restrict__ Pv0,
                        const float* __restrict__ Ev0, const float* __restrict__ Iv0,
                        float* __restrict__ S, _Float16* __restrict__ P16)
{
    const int n = blockIdx.x * blockDim.x + threadIdx.x;
    if (n < NN) {
        S[0 * NN + n] = P0[n];  S[1 * NN + n] = E0[n];  S[2 * NN + n] = I0[n];
        S[3 * NN + n] = Pv0[n]; S[4 * NN + n] = Ev0[n]; S[5 * NN + n] = Iv0[n];
        P16[n] = (_Float16)P0[n];
    }
}

// ---------------------------------------------------------------------------
// One Euler step. Block = 16 rows (one row tile), 8 waves split K=4096.
// y = W @ P via v_wmma_f32_16x16x32_f16 with P replicated across B columns.
// Then threads 0..15 run the Jansen-Rit pointwise update for their node.
// ---------------------------------------------------------------------------
__global__ void __launch_bounds__(256) jr_step(const _Float16* __restrict__ packedW,
                                               const float* __restrict__ rowsum,
                                               const float* __restrict__ scal,
                                               const float* __restrict__ params,
                                               const float* __restrict__ u,
                                               const float* __restrict__ noise,
                                               const float* __restrict__ Scur,
                                               float* __restrict__ Snxt,
                                               const _Float16* __restrict__ P16cur,
                                               _Float16* __restrict__ P16nxt,
                                               float* __restrict__ out,
                                               int t)
{
    __shared__ float red[8 * 16];
    const int tid  = threadIdx.x;
    const int wave = tid >> 5;
    const int lane = tid & 31;
    const int m    = blockIdx.x;           // row tile index (0..255)

    v8f acc = {};
    const int kb0 = wave * 16;             // this wave's first K-tile
    const _Float16* wbase = packedW + ((size_t)m * 128 + kb0) * 512 + lane * 16;
    const int bsel = (lane >> 4) << 4;     // lane<16 -> K 0..15, lane>=16 -> K 16..31
    #pragma unroll
    for (int i = 0; i < 16; ++i) {
        v16h a = *(const v16h*)(wbase + (size_t)i * 512);
        v16h b = *(const v16h*)(P16cur + (kb0 + i) * 32 + bsel);  // broadcast P slice
        acc = __builtin_amdgcn_wmma_f32_16x16x32_f16(false, a, false, b,
                                                     (short)0, acc, false, false);
    }
    // D layout: VGPR j, lanes 0-15 -> M=j, lanes 16-31 -> M=8+j; all N columns equal.
    if (lane == 0) {
        #pragma unroll
        for (int j = 0; j < 8; ++j) red[wave * 16 + j] = acc[j];
    } else if (lane == 16) {
        #pragma unroll
        for (int j = 0; j < 8; ++j) red[wave * 16 + 8 + j] = acc[j];
    }
    __syncthreads();

    if (tid < 16) {
        float y = 0.f;
        #pragma unroll
        for (int w = 0; w < 8; ++w) y += red[w * 16 + tid];

        const int n = m * 16 + tid;
        const float Ap = params[0],  a   = params[1],  Bp = params[2],  b  = params[3];
        const float vmax = params[4], v0 = params[5],  rr = params[6];
        const float c1 = params[7],  c2  = params[8],  c3 = params[9],  c4 = params[10];
        const float std_in = params[11], kk = params[12], ki = params[13];
        const float kE = params[14], kI = params[15], g = params[17];
        const float inv_norm = scal[0];

        const float P  = Scur[0 * NN + n], E  = Scur[1 * NN + n], I  = Scur[2 * NN + n];
        const float Pv = Scur[3 * NN + n], Ev = Scur[4 * NN + n], Iv = Scur[5 * NN + n];
        const float ut = u[(size_t)t * NN + n];
        const float nP = noise[((size_t)t * 3 + 0) * NN + n];
        const float nE = noise[((size_t)t * 3 + 1) * NN + n];
        const float nI = noise[((size_t)t * 3 + 2) * NN + n];

        const float lc = g * (y * inv_norm - rowsum[n] * inv_norm * P);
        auto S = [&](float x) { return vmax / (1.0f + expf(rr * (v0 - x))); };

        float rP = kk * ki * ut + std_in * nP + S(E - I);
        float rE = kE + std_in * nE + lc + c2 * S(c1 * P);
        float rI = kI + std_in * nI + c4 * S(c3 * P);
        rP = boundedf(rP, JR_IN_CLAMP);
        rE = boundedf(rE, JR_IN_CLAMP);
        rI = boundedf(rI, JR_IN_CLAMP);

        float Pn  = P + JR_DT * Pv;
        float En  = E + JR_DT * Ev;
        float In  = I + JR_DT * Iv;
        float Pvn = Pv + JR_DT * (Ap * a * rP - 2.0f * a * Pv - a * a * P);
        float Evn = Ev + JR_DT * (Ap * a * rE - 2.0f * a * Ev - a * a * E);
        float Ivn = Iv + JR_DT * (Bp * b * rI - 2.0f * b * Iv - b * b * I);

        Pn  = boundedf(Pn,  JR_ST_CLAMP);
        En  = boundedf(En,  JR_ST_CLAMP);
        In  = boundedf(In,  JR_ST_CLAMP);
        Pvn = boundedf(Pvn, JR_ST_CLAMP);
        Evn = boundedf(Evn, JR_ST_CLAMP);
        Ivn = boundedf(Ivn, JR_ST_CLAMP);

        Snxt[0 * NN + n] = Pn;  Snxt[1 * NN + n] = En;  Snxt[2 * NN + n] = In;
        Snxt[3 * NN + n] = Pvn; Snxt[4 * NN + n] = Evn; Snxt[5 * NN + n] = Ivn;
        P16nxt[n] = (_Float16)Pn;
        out[(size_t)t * NN + n] = Pn;
    }
}

// ---------------------------------------------------------------------------
// Host side: prep once, then T sequential steps (ping-pong state buffers).
// Workspace layout (bytes), all 32B-aligned:
//   [0)            packed W f16 : 4096*4096*2 = 33554432
//   [33554432)     rowsum  f32  : 16384
//   [33570816)     sqpart  f32  : 16384
//   [33587200)     scalars f32  : 256   (scal[0] = 1/||W||_F)
//   [33587456)     state buf 0  : 98304 (6 * 4096 f32)
//   [33685760)     state buf 1  : 98304
//   [33784064)     P16 buf 0    : 8192
//   [33792256)     P16 buf 1    : 8192      total ~33.8 MB
// ---------------------------------------------------------------------------
extern "C" void kernel_launch(void* const* d_in, const int* in_sizes, int n_in,
                              void* d_out, int out_size, void* d_ws, size_t ws_size,
                              hipStream_t stream)
{
    const float* sc     = (const float*)d_in[0];
    const float* u      = (const float*)d_in[1];
    const float* noise  = (const float*)d_in[2];
    const float* params = (const float*)d_in[3];
    const float* P0  = (const float*)d_in[4];
    const float* E0  = (const float*)d_in[5];
    const float* I0  = (const float*)d_in[6];
    const float* Pv0 = (const float*)d_in[7];
    const float* Ev0 = (const float*)d_in[8];
    const float* Iv0 = (const float*)d_in[9];
    const int T = in_sizes[1] / NN;

    char* ws = (char*)d_ws;
    _Float16* packedW = (_Float16*)(ws + 0);
    float*    rowsum  = (float*)(ws + 33554432);
    float*    sqpart  = (float*)(ws + 33570816);
    float*    scal    = (float*)(ws + 33587200);
    float*    state0  = (float*)(ws + 33587456);
    float*    state1  = (float*)(ws + 33685760);
    _Float16* p16a    = (_Float16*)(ws + 33784064);
    _Float16* p16b    = (_Float16*)(ws + 33792256);

    jr_prep_w<<<NN, 256, 0, stream>>>(sc, params, packedW, rowsum, sqpart);
    jr_norm<<<1, 256, 0, stream>>>(sqpart, scal);
    jr_init<<<NN / 256, 256, 0, stream>>>(P0, E0, I0, Pv0, Ev0, Iv0, state0, p16a);

    for (int t = 0; t < T; ++t) {
        const float*    Sc = (t & 1) ? state1 : state0;
        float*          Sn = (t & 1) ? state0 : state1;
        const _Float16* Pc = (t & 1) ? p16b : p16a;
        _Float16*       Pn = (t & 1) ? p16a : p16b;
        jr_step<<<NN / 16, 256, 0, stream>>>(packedW, rowsum, scal, params, u, noise,
                                             Sc, Sn, Pc, Pn, (float*)d_out, t);
    }
}
